// MoELayer_48636209660458
// MI455X (gfx1250) — matrix operand order repaired and checked by reading, not compiled, not run on hardware
//
#include <hip/hip_runtime.h>
#include <hip/hip_bf16.h>
#include <math.h>

typedef __bf16 bf16_t;
typedef __attribute__((ext_vector_type(16))) __bf16 v16bf;
typedef __attribute__((ext_vector_type(8)))  float  v8f;

#define N_TOK 8192
#define DIM   1024
#define FFN   4096
#define NE    8
#define TOPK  2
#define M_TILE 64
#define MT     (M_TILE / 16)         // 4 M-subtiles
#define F_CHUNK 128
#define MAX_TILES (N_TOK / M_TILE)   // 128

// Dynamic LDS layout (bytes): xs 128KB, hs 16KB, metadata
#define SM_XS   0
#define SM_HS   (M_TILE * DIM * 2)
#define SM_TK   (SM_HS + M_TILE * F_CHUNK * 2)
#define SM_GT   (SM_TK + M_TILE * 4)
#define SM_VLD  (SM_GT + M_TILE * 4)
#define SM_BYTES (SM_VLD + M_TILE * 4 + 128)

// ---------------------------------------------------------------------------
// Fragment loaders (contiguous k order, same for A and B, wave32)
// A 16x32 bf16: lane L -> row M = L&15; lanes 0-15 hold K 0..15 (v[0..15]);
// lanes 16-31 hold K 16..31.
__device__ __forceinline__ v16bf load_a_frag(const bf16_t* __restrict__ row,
                                             int k_base, int lane) {
  int koff = k_base + ((lane >> 4) << 4);
  v16bf a;
#pragma unroll
  for (int j = 0; j < 8; ++j) {
    a[j]     = row[koff + j];
    a[j + 8] = row[koff + 8 + j];
  }
  return a;
}

// B 32x16 bf16 from row-major f32 matrix (ld = row stride):
// lane L -> col N = L&15; lanes 0-15 hold K 0..15, lanes 16-31 hold K 16..31.
__device__ __forceinline__ v16bf load_b_frag(const float* __restrict__ base,
                                             int ld, int k_base, int col_base,
                                             int lane) {
  int col  = col_base + (lane & 15);
  int krow = k_base + ((lane >> 4) << 4);
  const float* p = base + (size_t)krow * ld + col;
  v16bf b;
#pragma unroll
  for (int j = 0; j < 16; ++j) b[j] = (bf16_t)p[(size_t)j * ld];
  return b;
}

// ---------------------------------------------------------------------------
__global__ void zero_counts_kernel(int* counts) {
  if (threadIdx.x < NE) counts[threadIdx.x] = 0;
}

__global__ void convert_x_kernel(const float* __restrict__ x,
                                 bf16_t* __restrict__ xb) {
  size_t i = (size_t)blockIdx.x * 256 + threadIdx.x;
  if (i < (size_t)N_TOK * DIM) xb[i] = (bf16_t)x[i];
}

// One wave per token: logits = x_row @ w_gate, softmax, top-2, append to lists.
__global__ __launch_bounds__(128) void gating_kernel(
    const float* __restrict__ x, const float* __restrict__ wg,
    int* __restrict__ counts, int* __restrict__ list,
    float* __restrict__ gates) {
  int lane = threadIdx.x & 31;
  int n = blockIdx.x * 4 + (threadIdx.x >> 5);
  if (n >= N_TOK) return;

  float p[NE];
#pragma unroll
  for (int e = 0; e < NE; ++e) p[e] = 0.0f;

  for (int d = lane; d < DIM; d += 32) {
    float xv = x[(size_t)n * DIM + d];
#pragma unroll
    for (int e = 0; e < NE; ++e) p[e] += xv * wg[d * NE + e];
  }
#pragma unroll
  for (int e = 0; e < NE; ++e)
    for (int off = 16; off > 0; off >>= 1)
      p[e] += __shfl_xor(p[e], off, 32);

  if (lane == 0) {
    float mx = p[0];
#pragma unroll
    for (int e = 1; e < NE; ++e) mx = fmaxf(mx, p[e]);
    float pr[NE]; float sum = 0.0f;
#pragma unroll
    for (int e = 0; e < NE; ++e) { pr[e] = __expf(p[e] - mx); sum += pr[e]; }
    float inv = 1.0f / sum;
#pragma unroll
    for (int e = 0; e < NE; ++e) pr[e] *= inv;

    int i0 = 0; float v0 = pr[0];
#pragma unroll
    for (int e = 1; e < NE; ++e) if (pr[e] > v0) { v0 = pr[e]; i0 = e; }
    int i1 = -1; float v1 = -1.0f;
#pragma unroll
    for (int e = 0; e < NE; ++e)
      if (e != i0 && pr[e] > v1) { v1 = pr[e]; i1 = e; }

    float denom = v0 + v1 + 1e-9f;
    float g0 = v0 / denom, g1 = v1 / denom;

    int s0 = atomicAdd(&counts[i0], 1);
    list[i0 * N_TOK + s0] = n * TOPK + 0;
    gates[n * TOPK + 0] = g0;
    int s1 = atomicAdd(&counts[i1], 1);
    list[i1 * N_TOK + s1] = n * TOPK + 1;
    gates[n * TOPK + 1] = g1;
  }
}

// Fused expert kernel: block = (expert, 64-token tile), 256 threads = 8 waves.
// Wave w owns output cols [w*128, w*128+128) for all 64 rows (4 M-subtiles).
// For each 128-wide F chunk:
//   GEMM1: h[64x128] = relu(x_tile[64x1024] @ W1_e[:,chunk])  (wave w -> 16 cols)
//   GEMM2: acc[64x128] += h @ W2_e[chunk, wave cols]
// Each B fragment feeds 4 WMMAs (all M-subtiles) -> ~4 vmem dwords per WMMA.
__global__ __launch_bounds__(256) void moe_expert_kernel(
    const bf16_t* __restrict__ xb, const float* __restrict__ W1,
    const float* __restrict__ W2, const int* __restrict__ counts,
    const int* __restrict__ list, const float* __restrict__ gates,
    float* __restrict__ out_slot) {
  int e    = blockIdx.x >> 7;             // / MAX_TILES
  int tile = blockIdx.x & (MAX_TILES - 1);
  int cnt  = counts[e];
  if (tile * M_TILE >= cnt) return;

  extern __shared__ char smem[];
  bf16_t* xs  = (bf16_t*)(smem + SM_XS);   // [M_TILE][DIM]
  bf16_t* hs  = (bf16_t*)(smem + SM_HS);   // [M_TILE][F_CHUNK]
  int*    tk  = (int*)(smem + SM_TK);
  float*  gt  = (float*)(smem + SM_GT);
  int*    vld = (int*)(smem + SM_VLD);

  int tid  = threadIdx.x;
  int lane = tid & 31;
  int w    = tid >> 5;

  if (tid < M_TILE) {
    int idx = tile * M_TILE + tid;
    if (idx < cnt) {
      int v = list[e * N_TOK + idx];
      tk[tid] = v; gt[tid] = gates[v]; vld[tid] = 1;
    } else {
      tk[tid] = 0; gt[tid] = 0.0f; vld[tid] = 0;
    }
  }
  __syncthreads();

  for (int i = tid; i < M_TILE * DIM; i += 256) {
    int m = i >> 10, d = i & (DIM - 1);
    xs[m * DIM + d] = vld[m] ? xb[(size_t)(tk[m] >> 1) * DIM + d]
                             : (bf16_t)0.0f;
  }
  __syncthreads();

  const float* w1e = W1 + (size_t)e * DIM * FFN;
  const float* w2e = W2 + (size_t)e * FFN * DIM;

  v8f acc[MT][8];
#pragma unroll
  for (int mt = 0; mt < MT; ++mt)
#pragma unroll
    for (int t = 0; t < 8; ++t) acc[mt][t] = (v8f){};

  const bf16_t* xrow[MT];
  const bf16_t* hrow[MT];
#pragma unroll
  for (int mt = 0; mt < MT; ++mt) {
    xrow[mt] = xs + (size_t)(mt * 16 + (lane & 15)) * DIM;
    hrow[mt] = hs + (size_t)(mt * 16 + (lane & 15)) * F_CHUNK;
  }
  int mb = (lane >> 4) << 3;   // C-fragment row base (M = r + 8*(lane/16))
  int nn = lane & 15;          // C-fragment column

  for (int f0 = 0; f0 < FFN; f0 += F_CHUNK) {
    // Pull this chunk's W2 slice (used by GEMM2 below) toward the caches
    // while GEMM1 computes: emits global_prefetch_b8.
    for (int r = lane; r < F_CHUNK; r += 32)
      __builtin_prefetch(w2e + (size_t)(f0 + r) * DIM + w * 128, 0, 1);

    // ---- GEMM1: wave w computes h cols [w*16, w*16+16) over K = D
    v8f a1[MT];
#pragma unroll
    for (int mt = 0; mt < MT; ++mt) a1[mt] = (v8f){};
    int colb1 = f0 + w * 16;
    for (int kb = 0; kb < DIM; kb += 32) {
      v16bf B = load_b_frag(w1e, FFN, kb, colb1, lane);
#pragma unroll
      for (int mt = 0; mt < MT; ++mt) {
        v16bf A = load_a_frag(xrow[mt], kb, lane);
        a1[mt] = __builtin_amdgcn_wmma_f32_16x16x32_bf16(
            false, A, false, B, (short)0, a1[mt], false, false);
      }
    }
    __syncthreads();  // WAR: previous chunk's GEMM2 reads of hs done
#pragma unroll
    for (int mt = 0; mt < MT; ++mt)
#pragma unroll
      for (int r = 0; r < 8; ++r) {
        float hv = a1[mt][r] > 0.0f ? a1[mt][r] : 0.0f;  // ReLU
        hs[(size_t)(mt * 16 + mb + r) * F_CHUNK + w * 16 + nn] = (bf16_t)hv;
      }
    __syncthreads();  // h visible to all waves

    // ---- GEMM2: acc[64 x 128] += h[64 x 128] @ W2[f0:f0+128, wave cols]
    for (int kb = 0; kb < F_CHUNK; kb += 32) {
      v16bf A[MT];
#pragma unroll
      for (int mt = 0; mt < MT; ++mt) A[mt] = load_a_frag(hrow[mt], kb, lane);
#pragma unroll
      for (int t = 0; t < 8; ++t) {
        v16bf B = load_b_frag(w2e, DIM, f0 + kb, w * 128 + t * 16, lane);
#pragma unroll
        for (int mt = 0; mt < MT; ++mt)
          acc[mt][t] = __builtin_amdgcn_wmma_f32_16x16x32_bf16(
              false, A[mt], false, B, (short)0, acc[mt][t], false, false);
      }
    }
  }

  // ---- write gate-scaled contribution to per-(token,slot) buffer
#pragma unroll
  for (int mt = 0; mt < MT; ++mt)
#pragma unroll
    for (int t = 0; t < 8; ++t) {
      int col = w * 128 + t * 16 + nn;
#pragma unroll
      for (int r = 0; r < 8; ++r) {
        int m = mt * 16 + mb + r;
        if (vld[m])
          out_slot[(size_t)tk[m] * DIM + col] = acc[mt][t][r] * gt[m];
      }
    }
}

__global__ void combine_kernel(const float* __restrict__ os,
                               float* __restrict__ y) {
  size_t i = (size_t)blockIdx.x * 256 + threadIdx.x;
  if (i < (size_t)N_TOK * DIM) {
    size_t n = i >> 10;
    int d = (int)(i & (DIM - 1));
    y[i] = os[(n * 2 + 0) * DIM + d] + os[(n * 2 + 1) * DIM + d];
  }
}

// ---------------------------------------------------------------------------
extern "C" void kernel_launch(void* const* d_in, const int* in_sizes, int n_in,
                              void* d_out, int out_size, void* d_ws,
                              size_t ws_size, hipStream_t stream) {
  const float* x  = (const float*)d_in[0];   // [N, D]
  const float* wg = (const float*)d_in[1];   // [D, E]
  const float* W1 = (const float*)d_in[2];   // [E, D, F]
  const float* W2 = (const float*)d_in[3];   // [E, F, D]
  float* y = (float*)d_out;                  // [N, D]

  char* ws = (char*)d_ws;
  size_t off = 0;
  int* counts = (int*)(ws + off);           off += 256;
  int* list   = (int*)(ws + off);           off += (size_t)NE * N_TOK * 4;
  float* gates = (float*)(ws + off);        off += (size_t)N_TOK * TOPK * 4;
  bf16_t* xb  = (bf16_t*)(ws + off);        off += (size_t)N_TOK * DIM * 2;
  float* out_slot = (float*)(ws + off);     off += (size_t)N_TOK * TOPK * DIM * 4;
  (void)ws_size; (void)in_sizes; (void)n_in; (void)out_size;

  zero_counts_kernel<<<1, 32, 0, stream>>>(counts);

  int cvt_blocks = (int)(((size_t)N_TOK * DIM + 255) / 256);
  convert_x_kernel<<<cvt_blocks, 256, 0, stream>>>(x, xb);

  gating_kernel<<<N_TOK / 4, 128, 0, stream>>>(x, wg, counts, list, gates);

  moe_expert_kernel<<<NE * MAX_TILES, 256, SM_BYTES, stream>>>(
      xb, W1, W2, counts, list, gates, out_slot);

  int cmb_blocks = (int)(((size_t)N_TOK * DIM + 255) / 256);
  combine_kernel<<<cmb_blocks, 256, 0, stream>>>(out_slot, y);
}
